// BIOTEncoder_8873402433856
// MI455X (gfx1250) — matrix operand order repaired and checked by reading, not compiled, not run on hardware
//
#include <hip/hip_runtime.h>
#include <hip/hip_bf16.h>
#include <math.h>

typedef __attribute__((ext_vector_type(16))) _Float16 v16h;
typedef __attribute__((ext_vector_type(8)))  float    v8f;

#define BATCH 256
#define NCH   16
#define TLEN  2000
#define NFFT  200
#define HOPSZ 100
#define NFR   19
#define NFREQ 101
#define EMB   256
#define NHEAD 8
#define DHEAD 32
#define DEPTH 4
#define FF    1024
#define SEQ   (NCH * NFR)      // 304
#define ROWS  (BATCH * SEQ)    // 77824
#define LN_EPS 1e-5f
#define QSCALE 0.17677669529663687f   // dh^-0.5

// ---------------- WMMA helpers (CDNA5 gfx1250, wave32) ----------------

__device__ __forceinline__ v8f wmma16x16x32(const v16h& a, const v16h& b, const v8f& c) {
  // D = A(16x32 f16) * B(32x16 f16) + C(16x16 f32)
  return __builtin_amdgcn_wmma_f32_16x16x32_f16(false, a, false, b, (short)0, c, false, false);
}

// A fragment: 16x32 f16 tile from row-major source, leading dim lda.
// ISA layout: lane holds row m=lane&15; element e -> k = (e>>3)*16 + (lane>>4)*8 + (e&7)
// (two contiguous 8-element runs per lane -> compiler emits ds_load_b128 pairs)
__device__ __forceinline__ v16h load_a_frag(const _Float16* __restrict__ A, int lda,
                                            int k0, int lane) {
  v16h a;
  const int half = lane >> 4, m = lane & 15;
#pragma unroll
  for (int e = 0; e < 16; ++e) {
    int k = k0 + ((e >> 3) << 4) + (half << 3) + (e & 7);
    a[e] = A[(size_t)m * lda + k];
  }
  return a;
}

// B fragment from PRE-PACKED weights: lane's 16 operand f16s are contiguous (32B).
// Packed layout: frag (kt, nt) base = ((nt*ktiles + kt)*32 + lane)*16 elements.
__device__ __forceinline__ v16h load_b_packed(const _Float16* __restrict__ Wp,
                                              int ktiles, int kt, int nt, int lane) {
  const v16h* p = (const v16h*)(Wp + (((size_t)nt * ktiles + kt) * 32 + (size_t)lane) * 16);
  return *p;
}

// Async global -> LDS staging of a 16xEMB f16 tile (8 KB): 512 threads x 16 B.
// Uses CDNA5 GLOBAL_LOAD_ASYNC_TO_LDS_B128 (ASYNCcnt), GVS addressing.
__device__ __forceinline__ void async_stage_16x256(const _Float16* __restrict__ gbase,
                                                   int row0, _Float16* lds, int tid) {
  unsigned int ldsa = (unsigned int)(unsigned long long)lds + (unsigned int)tid * 16u;
  unsigned int goff = (unsigned int)((((unsigned int)(row0 + (tid >> 5)) * EMB)
                                      + (unsigned int)(tid & 31) * 8u) * 2u);
  asm volatile("global_load_async_to_lds_b128 %0, %1, %2"
               :: "v"(ldsa), "v"(goff), "s"(gbase) : "memory");
}

__device__ __forceinline__ void async_wait() {
  asm volatile("s_wait_asynccnt 0x0" ::: "memory");
}

// ---------------- prep kernels ----------------

__global__ void build_basis_kernel(float* __restrict__ cosb, float* __restrict__ sinb) {
  int i = blockIdx.x * blockDim.x + threadIdx.x;
  if (i >= NFREQ * NFFT) return;
  int k = i / NFFT, n = i % NFFT;
  const float PI = 3.14159265358979323846f;
  float win = 0.5f * (1.f - cosf(2.f * PI * (float)n / (float)NFFT));
  float ang = 2.f * PI * (float)k * (float)n / (float)NFFT;
  cosb[i] = win * cosf(ang);
  sinb[i] = win * sinf(ang);
}

// Pack (DEPTH, K, N) f32 weights into per-layer WMMA-B fragment order (f16).
__global__ void pack_weight_kernel(const float* __restrict__ src, _Float16* __restrict__ dst,
                                   int K, int N, int total) {
  const int ktiles = K >> 5;
  const int layerKN = K * N;
  for (int i = blockIdx.x * blockDim.x + threadIdx.x; i < total; i += gridDim.x * blockDim.x) {
    int l = i / layerKN;
    int j = i - l * layerKN;
    int e = j & 15;
    int lane = (j >> 4) & 31;
    int tile = j >> 9;
    int kt = tile % ktiles;
    int nt = tile / ktiles;
    int k = kt * 32 + ((e >> 3) << 4) + ((lane >> 4) << 3) + (e & 7);
    int n = nt * 16 + (lane & 15);
    dst[i] = (_Float16)src[(size_t)l * layerKN + (size_t)k * N + n];
  }
}

// ---------------- STFT + embedding ----------------
// grid (NFR, NCH, BATCH), block 128
__global__ void __launch_bounds__(128) stft_embed_kernel(
    const float* __restrict__ x, const int* __restrict__ noff,
    const float* __restrict__ projW, const float* __restrict__ projB,
    const float* __restrict__ chan, const float* __restrict__ cosb,
    const float* __restrict__ sinb, float* __restrict__ h) {
  __shared__ float frame[NFFT];
  __shared__ float spec[NFREQ + 3];
  const int fr = blockIdx.x, c = blockIdx.y, b = blockIdx.z;
  const int t = threadIdx.x;
  const float* xp = x + ((size_t)b * NCH + c) * TLEN + fr * HOPSZ;
  for (int i = t; i < NFFT; i += 128) frame[i] = xp[i];
  __syncthreads();
  for (int k = t; k < NFREQ; k += 128) {
    float re = 0.f, im = 0.f;
    const float* cb = cosb + (size_t)k * NFFT;
    const float* sb = sinb + (size_t)k * NFFT;
    for (int n = 0; n < NFFT; ++n) { re += frame[n] * cb[n]; im += frame[n] * sb[n]; }
    spec[k] = sqrtf(re * re + im * im);
  }
  __syncthreads();
  // dynamic_slice start clamped to [0, C - C] == 0
  int start = noff[0];
  if (start < 0) start = 0;
  if (start > NCH - NCH) start = NCH - NCH;
  const float* ct = chan + (size_t)(start + c) * EMB;
  const size_t row = (size_t)b * SEQ + (size_t)c * NFR + fr;
  const float kexp = -9.210340371976184f / (float)EMB;  // -ln(10000)/E
  for (int e = t; e < EMB; e += 128) {
    float acc = projB[e];
    for (int k = 0; k < NFREQ; ++k) acc += spec[k] * projW[(size_t)k * EMB + e];
    acc += ct[e];
    float df = __expf((float)(e & ~1) * kexp);
    float ang = (float)fr * df;
    acc += (e & 1) ? cosf(ang) : sinf(ang);
    h[row * EMB + e] = acc;
  }
}

// ---------------- LayerNorm -> f16 ----------------
// grid ROWS, block 256 (one row per block)
__global__ void __launch_bounds__(256) ln_kernel(
    const float* __restrict__ h, const float* __restrict__ g,
    const float* __restrict__ bta, _Float16* __restrict__ y) {
  __shared__ float red[256];
  const int row = blockIdx.x, t = threadIdx.x;
  float xv = h[(size_t)row * EMB + t];
  red[t] = xv;
  __syncthreads();
  for (int s = 128; s > 0; s >>= 1) { if (t < s) red[t] += red[t + s]; __syncthreads(); }
  float mu = red[0] * (1.f / EMB);
  __syncthreads();
  float dx = xv - mu;
  red[t] = dx * dx;
  __syncthreads();
  for (int s = 128; s > 0; s >>= 1) { if (t < s) red[t] += red[t + s]; __syncthreads(); }
  float var = red[0] * (1.f / EMB);
  y[(size_t)row * EMB + t] = (_Float16)(dx * rsqrtf(var + LN_EPS) * g[t] + bta[t]);
}

// ---------------- fused QKV GEMM (+ q feature softmax) ----------------
// grid (ROWS/16, 3), block 512 = 16 waves; wave w -> 16x16 tile at cols [16w,16w+16)
__global__ void __launch_bounds__(512) gemm_qkv_kernel(
    const _Float16* __restrict__ y, const _Float16* __restrict__ wpk,
    _Float16* __restrict__ q16, float* __restrict__ k32, _Float16* __restrict__ v16,
    int layer) {
  __shared__ _Float16 As[16 * EMB];   // 8 KB
  __shared__ float Ds[16 * EMB];      // 16 KB
  const int tid = threadIdx.x, lane = tid & 31, wave = tid >> 5;
  const int row0 = blockIdx.x * 16, sel = blockIdx.y;
  const _Float16* W = wpk + (size_t)sel * (4 * EMB * EMB) + (size_t)layer * EMB * EMB;
  async_stage_16x256(y, row0, As, tid);
  async_wait();
  __syncthreads();
  const int col0 = wave * 16;
  v8f acc = {};
  for (int ks = 0; ks < EMB / 32; ++ks) {
    v16h a = load_a_frag(As, EMB, ks * 32, lane);
    v16h b = load_b_packed(W, EMB / 32, ks, wave, lane);
    acc = wmma16x16x32(a, b, acc);
  }
  {
    const int half = lane >> 4, n = col0 + (lane & 15);
#pragma unroll
    for (int r = 0; r < 8; ++r) Ds[(half * 8 + r) * EMB + n] = acc[r];
  }
  __syncthreads();
  if (sel == 0) {
    // softmax over each 32-wide head chunk; wave w owns row w, lane = feature in chunk
    const int row = wave, d = lane;
    for (int chunk = 0; chunk < NHEAD; ++chunk) {
      float xv = Ds[row * EMB + chunk * 32 + d];
      float mx = xv;
      for (int o = 16; o > 0; o >>= 1) mx = fmaxf(mx, __shfl_xor(mx, o, 32));
      float ev = __expf(xv - mx);
      float sm = ev;
      for (int o = 16; o > 0; o >>= 1) sm += __shfl_xor(sm, o, 32);
      q16[(size_t)(row0 + row) * EMB + chunk * 32 + d] = (_Float16)(ev / sm * QSCALE);
    }
  } else if (sel == 1) {
    for (int i = tid; i < 16 * EMB; i += 512)
      k32[(size_t)(row0 + (i >> 8)) * EMB + (i & 255)] = Ds[i];
  } else {
    for (int i = tid; i < 16 * EMB; i += 512)
      v16[(size_t)(row0 + (i >> 8)) * EMB + (i & 255)] = (_Float16)Ds[i];
  }
}

// ---------------- k sequence softmax (axis = seq) ----------------
// grid BATCH*NHEAD, block 256: 32 cols x 8 row-groups
__global__ void __launch_bounds__(256) ksoftmax_kernel(
    const float* __restrict__ k32, _Float16* __restrict__ k16) {
  __shared__ float part[8][32];
  __shared__ float cmax[32];
  __shared__ float csum[32];
  const int bh = blockIdx.x, b = bh >> 3, hh = bh & 7;
  const int t = threadIdx.x, d = t & 31, gq = t >> 5;
  const size_t base = (size_t)b * SEQ * EMB + hh * DHEAD + d;
  float mx = -3.4e38f;
  for (int n = gq; n < SEQ; n += 8) mx = fmaxf(mx, k32[base + (size_t)n * EMB]);
  part[gq][d] = mx;
  __syncthreads();
  if (t < 32) { float m = part[0][t]; for (int i = 1; i < 8; ++i) m = fmaxf(m, part[i][t]); cmax[t] = m; }
  __syncthreads();
  const float cm = cmax[d];
  float sm = 0.f;
  for (int n = gq; n < SEQ; n += 8) sm += __expf(k32[base + (size_t)n * EMB] - cm);
  part[gq][d] = sm;
  __syncthreads();
  if (t < 32) { float s = 0.f; for (int i = 0; i < 8; ++i) s += part[i][t]; csum[t] = s; }
  __syncthreads();
  const float inv = 1.f / csum[d];
  for (int n = gq; n < SEQ; n += 8)
    k16[base + (size_t)n * EMB] = (_Float16)(__expf(k32[base + (size_t)n * EMB] - cm) * inv);
}

// ---------------- ctx = k^T v (32x32 per (b,h), K = SEQ padded to 320) ----------------
// grid BATCH*NHEAD, block 128 = 4 waves -> 4 16x16 tiles
__global__ void __launch_bounds__(128) ctx_kernel(
    const _Float16* __restrict__ k16, const _Float16* __restrict__ v16,
    _Float16* __restrict__ ctx) {
  const int bh = blockIdx.x, b = bh >> 3, hh = bh & 7;
  const int tid = threadIdx.x, lane = tid & 31, wave = tid >> 5;
  const int m0 = (wave >> 1) * 16, n0c = (wave & 1) * 16;
  const int half = lane >> 4;
  const _Float16 zero = (_Float16)0.f;
  v8f acc = {};
  for (int step = 0; step < 10; ++step) {
    const int kbase = step * 32;
    v16h a, bf;
#pragma unroll
    for (int e = 0; e < 16; ++e) {
      int kk = kbase + ((e >> 3) << 4) + (half << 3) + (e & 7);
      int m = m0 + (lane & 15);
      a[e] = (kk < SEQ) ? k16[((size_t)b * SEQ + kk) * EMB + hh * DHEAD + m] : zero;
    }
#pragma unroll
    for (int e = 0; e < 16; ++e) {
      int kk = kbase + ((e >> 3) << 4) + (half << 3) + (e & 7);
      int n = n0c + (lane & 15);
      bf[e] = (kk < SEQ) ? v16[((size_t)b * SEQ + kk) * EMB + hh * DHEAD + n] : zero;
    }
    acc = wmma16x16x32(a, bf, acc);
  }
  const int n = n0c + (lane & 15);
#pragma unroll
  for (int r = 0; r < 8; ++r)
    ctx[(size_t)bh * (DHEAD * DHEAD) + (m0 + half * 8 + r) * DHEAD + n] = (_Float16)acc[r];
}

// ---------------- o = q @ ctx ----------------
// grid (SEQ/16, BATCH), block 512 = 16 waves; wave w -> head w/2, d-half w&1; K = 32 (one WMMA)
__global__ void __launch_bounds__(512) attn_out_kernel(
    const _Float16* __restrict__ q16, const _Float16* __restrict__ ctx,
    _Float16* __restrict__ o16) {
  const int b = blockIdx.y, n0row = blockIdx.x * 16;
  const int tid = threadIdx.x, lane = tid & 31, wave = tid >> 5;
  const int head = wave >> 1, dsub = (wave & 1) * 16, half = lane >> 4;
  v16h a, bf;
#pragma unroll
  for (int e = 0; e < 16; ++e) {
    int kk = ((e >> 3) << 4) + (half << 3) + (e & 7);  // 0..31 = dh
    int m = lane & 15;
    a[e] = q16[((size_t)b * SEQ + n0row + m) * EMB + head * DHEAD + kk];
  }
#pragma unroll
  for (int e = 0; e < 16; ++e) {
    int kk = ((e >> 3) << 4) + (half << 3) + (e & 7);
    int n = lane & 15;
    bf[e] = ctx[(size_t)(b * NHEAD + head) * (DHEAD * DHEAD) + kk * DHEAD + dsub + n];
  }
  v8f acc = {};
  acc = wmma16x16x32(a, bf, acc);
  const int n = wave * 16 + (lane & 15);
#pragma unroll
  for (int r = 0; r < 8; ++r)
    o16[((size_t)b * SEQ + n0row + half * 8 + r) * EMB + n] = (_Float16)acc[r];
}

// ---------------- h += o @ Wo + bo ----------------
// grid ROWS/16, block 512
__global__ void __launch_bounds__(512) gemm_out_kernel(
    const _Float16* __restrict__ o16, const _Float16* __restrict__ wpk,
    const float* __restrict__ bo, float* __restrict__ h, int layer) {
  __shared__ _Float16 As[16 * EMB];
  const int tid = threadIdx.x, lane = tid & 31, wave = tid >> 5;
  const int row0 = blockIdx.x * 16;
  const _Float16* W = wpk + (size_t)(12 * EMB * EMB) + (size_t)layer * EMB * EMB;
  const float* bias = bo + (size_t)layer * EMB;
  async_stage_16x256(o16, row0, As, tid);
  async_wait();
  __syncthreads();
  const int col0 = wave * 16;
  v8f acc = {};
  for (int ks = 0; ks < EMB / 32; ++ks) {
    v16h a = load_a_frag(As, EMB, ks * 32, lane);
    v16h b = load_b_packed(W, EMB / 32, ks, wave, lane);
    acc = wmma16x16x32(a, b, acc);
  }
  const int half = lane >> 4, n = col0 + (lane & 15);
#pragma unroll
  for (int r = 0; r < 8; ++r) {
    size_t idx = (size_t)(row0 + half * 8 + r) * EMB + n;
    h[idx] += acc[r] + bias[n];
  }
}

// ---------------- fused FFN: h += gelu(y@W1+b1)@W2 + b2 ----------------
// grid ROWS/16, block 512; 40 KB LDS
__global__ void __launch_bounds__(512) ffn_kernel(
    const _Float16* __restrict__ y, const _Float16* __restrict__ wpk,
    const float* __restrict__ b1, const float* __restrict__ b2,
    float* __restrict__ h, int layer) {
  __shared__ _Float16 Ys[16 * EMB];  // 8 KB
  __shared__ _Float16 Ts[16 * FF];   // 32 KB
  const int tid = threadIdx.x, lane = tid & 31, wave = tid >> 5;
  const int row0 = blockIdx.x * 16;
  const _Float16* W1 = wpk + (size_t)(16 * EMB * EMB) + (size_t)layer * EMB * FF;
  const _Float16* W2 = wpk + (size_t)(16 * EMB * EMB) + (size_t)4 * EMB * FF + (size_t)layer * FF * EMB;
  const float* bb1 = b1 + (size_t)layer * FF;
  const float* bb2 = b2 + (size_t)layer * EMB;
  async_stage_16x256(y, row0, Ys, tid);
  async_wait();
  __syncthreads();
  const int half = lane >> 4;
  // phase 1: t = gelu(y@W1 + b1), 64 col tiles, 4 per wave
  for (int j = 0; j < 4; ++j) {
    const int nt = wave * 4 + j;
    const int col0 = nt * 16;
    v8f acc = {};
    for (int ks = 0; ks < EMB / 32; ++ks) {
      v16h a = load_a_frag(Ys, EMB, ks * 32, lane);
      v16h b = load_b_packed(W1, EMB / 32, ks, nt, lane);
      acc = wmma16x16x32(a, b, acc);
    }
    const int n = col0 + (lane & 15);
#pragma unroll
    for (int r = 0; r < 8; ++r) {
      float xv = acc[r] + bb1[n];
      float g = 0.5f * xv * (1.f + erff(xv * 0.70710678118654752f));
      Ts[(half * 8 + r) * FF + n] = (_Float16)g;
    }
  }
  __syncthreads();
  // phase 2: h += t@W2 + b2, 16 col tiles, 1 per wave, K = 1024
  {
    const int col0 = wave * 16;
    v8f acc = {};
    for (int ks = 0; ks < FF / 32; ++ks) {
      v16h a = load_a_frag(Ts, FF, ks * 32, lane);
      v16h b = load_b_packed(W2, FF / 32, ks, wave, lane);
      acc = wmma16x16x32(a, b, acc);
    }
    const int n = col0 + (lane & 15);
#pragma unroll
    for (int r = 0; r < 8; ++r) {
      size_t idx = (size_t)(row0 + half * 8 + r) * EMB + n;
      h[idx] += acc[r] + bb2[n];
    }
  }
}

// ---------------- outputs ----------------

__global__ void __launch_bounds__(256) mean_kernel(const float* __restrict__ h,
                                                   float* __restrict__ out) {
  const int b = blockIdx.x, e = threadIdx.x;
  float s = 0.f;
  for (int n = 0; n < SEQ; ++n) s += h[((size_t)b * SEQ + n) * EMB + e];
  out[(size_t)b * EMB + e] = s * (1.f / (float)SEQ);
}

__global__ void copy_kernel(const float* __restrict__ h, float* __restrict__ out, size_t n) {
  for (size_t i = (size_t)blockIdx.x * blockDim.x + threadIdx.x; i < n;
       i += (size_t)gridDim.x * blockDim.x)
    out[i] = h[i];
}

// ---------------- host ----------------

extern "C" void kernel_launch(void* const* d_in, const int* in_sizes, int n_in,
                              void* d_out, int out_size, void* d_ws, size_t ws_size,
                              hipStream_t stream) {
  (void)in_sizes; (void)n_in; (void)out_size; (void)ws_size;
  const float* x     = (const float*)d_in[0];
  const int*   noff  = (const int*)d_in[1];
  const float* projW = (const float*)d_in[2];
  const float* projB = (const float*)d_in[3];
  const float* chan  = (const float*)d_in[4];
  const float* Wq    = (const float*)d_in[5];
  const float* Wk    = (const float*)d_in[6];
  const float* Wv    = (const float*)d_in[7];
  const float* Wo    = (const float*)d_in[8];
  const float* bo    = (const float*)d_in[9];
  const float* ln1g  = (const float*)d_in[10];
  const float* ln1b  = (const float*)d_in[11];
  const float* ln2g  = (const float*)d_in[12];
  const float* ln2b  = (const float*)d_in[13];
  const float* W1    = (const float*)d_in[14];
  const float* b1    = (const float*)d_in[15];
  const float* W2    = (const float*)d_in[16];
  const float* b2    = (const float*)d_in[17];

  // workspace layout (bytes)
  constexpr size_t szH   = (size_t)ROWS * EMB * 4;   // 79,691,776
  constexpr size_t szF16 = (size_t)ROWS * EMB * 2;   // 39,845,888
  constexpr size_t szCtx = (size_t)BATCH * NHEAD * DHEAD * DHEAD * 2;
  constexpr size_t szW16 = (size_t)(16 * EMB * EMB + 8 * EMB * FF) * 2;  // 6,291,456
  constexpr size_t szBas = (size_t)NFREQ * NFFT * 4;

  char* ws = (char*)d_ws;
  size_t off = 0;
  float*     h    = (float*)(ws + off);     off += szH;
  _Float16*  y16  = (_Float16*)(ws + off);  off += szF16;
  _Float16*  q16  = (_Float16*)(ws + off);  off += szF16;
  float*     k32  = (float*)(ws + off);     off += szH;
  _Float16*  k16  = (_Float16*)(ws + off);  off += szF16;
  _Float16*  v16  = (_Float16*)(ws + off);  off += szF16;
  _Float16*  ctx  = (_Float16*)(ws + off);  off += szCtx;
  _Float16*  o16  = (_Float16*)(ws + off);  off += szF16;
  _Float16*  wpk  = (_Float16*)(ws + off);  off += szW16;
  float*     cosb = (float*)(ws + off);     off += szBas;
  float*     sinb = (float*)(ws + off);     off += szBas;

  // ---- prep: DFT basis + packed f16 weights (WMMA-B fragment order) ----
  build_basis_kernel<<<(NFREQ * NFFT + 255) / 256, 256, 0, stream>>>(cosb, sinb);
  pack_weight_kernel<<<512, 256, 0, stream>>>(Wq, wpk + (size_t)0 * 4 * EMB * EMB, EMB, EMB, 4 * EMB * EMB);
  pack_weight_kernel<<<512, 256, 0, stream>>>(Wk, wpk + (size_t)1 * 4 * EMB * EMB, EMB, EMB, 4 * EMB * EMB);
  pack_weight_kernel<<<512, 256, 0, stream>>>(Wv, wpk + (size_t)2 * 4 * EMB * EMB, EMB, EMB, 4 * EMB * EMB);
  pack_weight_kernel<<<512, 256, 0, stream>>>(Wo, wpk + (size_t)3 * 4 * EMB * EMB, EMB, EMB, 4 * EMB * EMB);
  pack_weight_kernel<<<1024, 256, 0, stream>>>(W1, wpk + (size_t)16 * EMB * EMB, EMB, FF, 4 * EMB * FF);
  pack_weight_kernel<<<1024, 256, 0, stream>>>(W2, wpk + (size_t)16 * EMB * EMB + (size_t)4 * EMB * FF,
                                               FF, EMB, 4 * FF * EMB);

  // ---- STFT + projection + tokens + PE -> h ----
  stft_embed_kernel<<<dim3(NFR, NCH, BATCH), 128, 0, stream>>>(
      x, noff, projW, projB, chan, cosb, sinb, h);

  // ---- transformer layers ----
  for (int l = 0; l < DEPTH; ++l) {
    ln_kernel<<<ROWS, 256, 0, stream>>>(h, ln1g + l * EMB, ln1b + l * EMB, y16);
    gemm_qkv_kernel<<<dim3(ROWS / 16, 3), 512, 0, stream>>>(y16, wpk, q16, k32, v16, l);
    ksoftmax_kernel<<<BATCH * NHEAD, 256, 0, stream>>>(k32, k16);
    ctx_kernel<<<BATCH * NHEAD, 128, 0, stream>>>(k16, v16, ctx);
    attn_out_kernel<<<dim3(SEQ / 16, BATCH), 512, 0, stream>>>(q16, ctx, o16);
    gemm_out_kernel<<<ROWS / 16, 512, 0, stream>>>(o16, wpk, bo, h, l);
    ln_kernel<<<ROWS, 256, 0, stream>>>(h, ln2g + l * EMB, ln2b + l * EMB, y16);
    ffn_kernel<<<ROWS / 16, 512, 0, stream>>>(y16, wpk, b1, b2, h, l);
  }

  // ---- outputs: mean over seq, then full h ----
  mean_kernel<<<BATCH, 256, 0, stream>>>(h, (float*)d_out);
  copy_kernel<<<4096, 256, 0, stream>>>(h, (float*)d_out + (size_t)BATCH * EMB,
                                        (size_t)ROWS * EMB);
}
